// GraphCrossnet_77635828842628
// MI455X (gfx1250) — matrix operand order repaired and checked by reading, not compiled, not run on hardware
//
#include <hip/hip_runtime.h>
#include <math.h>

// ---------------------------------------------------------------------------
// Types for CDNA5 WMMA (gfx1250, wave32)
// ---------------------------------------------------------------------------
typedef __attribute__((ext_vector_type(16))) __bf16 v16bf;
typedef __attribute__((ext_vector_type(8)))  float  v8f;

#define TPB 256

__device__ __forceinline__ unsigned short f2bf(float f) {
    union { float f; unsigned u; } v; v.f = f;
    unsigned r = v.u + 0x7FFFu + ((v.u >> 16) & 1u);   // round-to-nearest-even
    return (unsigned short)(r >> 16);
}

__device__ __forceinline__ float sigm(float x) { return 1.0f / (1.0f + expf(-x)); }

// ---------------------------------------------------------------------------
// Pack A (f32, optionally row/col-gathered) -> bf16 WMMA-A fragment blocks.
// Block (mt,kt) is 512 bf16 (1KB): element [lane*16+e] holds A[m,k] with
//   m = mt*16 + (lane&15),  k = kt*32 + e + 8*(e>=8) + 8*(lane>=16)
// ---------------------------------------------------------------------------
__global__ void k_pack_A(const float* __restrict__ A, unsigned short* __restrict__ Asw,
                         const int* __restrict__ ridx, const int* __restrict__ cidx,
                         int Mrows, int Kcols, int nMt, int nKt, int lda)
{
    long tid = (long)blockIdx.x * blockDim.x + threadIdx.x;
    long blk = tid >> 5;
    if (blk >= (long)nMt * nKt) return;
    int lane = (int)(tid & 31);
    int mt = (int)(blk / nKt), kt = (int)(blk % nKt);
    int gm = mt * 16 + (lane & 15);
    long rmap = -1;
    if (gm < Mrows) rmap = ridx ? (long)ridx[gm] : (long)gm;

    __attribute__((aligned(16))) unsigned short h[16];
#pragma unroll
    for (int e = 0; e < 16; ++e) {
        int kin = e + ((e >= 8) ? 8 : 0) + ((lane >= 16) ? 8 : 0);
        int gk = kt * 32 + kin;
        float v = 0.0f;
        if (rmap >= 0 && gk < Kcols) {
            long cm = cidx ? (long)cidx[gk] : (long)gk;
            v = A[rmap * (long)lda + cm];
        }
        h[e] = f2bf(v);
    }
    unsigned short* dst = Asw + blk * 512 + lane * 16;
    *(uint4*)(dst + 0) = *(const uint4*)(h + 0);
    *(uint4*)(dst + 8) = *(const uint4*)(h + 8);
}

// ---------------------------------------------------------------------------
// Pack X (f32 [rows,C], with optional inverse-scatter map: row r of the
// logical K x C matrix is X[inv[r]] or zero) -> bf16 WMMA-B fragment blocks.
// Block (kt,nt): element [lane*16+e] holds Xlogical[k,n] with
//   k = kt*32 + e + 8*(e>=8) + 8*(lane>=16),  n = nt*16 + (lane&15)
// ---------------------------------------------------------------------------
__global__ void k_pack_B(const float* __restrict__ X, unsigned short* __restrict__ Bsw,
                         int Krows, int C, int nKt, int nNt, const int* __restrict__ inv)
{
    long tid = (long)blockIdx.x * blockDim.x + threadIdx.x;
    long blk = tid >> 5;
    if (blk >= (long)nKt * nNt) return;
    int lane = (int)(tid & 31);
    int kt = (int)(blk / nNt), nt = (int)(blk % nNt);
    int n = nt * 16 + (lane & 15);

    __attribute__((aligned(16))) unsigned short h[16];
#pragma unroll
    for (int e = 0; e < 16; ++e) {
        int k = kt * 32 + e + ((e >= 8) ? 8 : 0) + ((lane >= 16) ? 8 : 0);
        float v = 0.0f;
        if (k < Krows) {
            int r = inv ? inv[k] : k;
            if (r >= 0) v = X[(long)r * C + n];
        }
        h[e] = f2bf(v);
    }
    unsigned short* dst = Bsw + (long)blk * 512 + lane * 16;
    *(uint4*)(dst + 0) = *(const uint4*)(h + 0);
    *(uint4*)(dst + 8) = *(const uint4*)(h + 8);
}

// ---------------------------------------------------------------------------
// GEMM with intra-workgroup INTERLEAVED split-K:
//   Y[M, NT*16] = A(swizzled bf16) @ B(swizzled bf16), f32 accumulate.
// One BLOCK (8 waves) per 16-row M tile; wave w accumulates k-tiles
// w, w+8, w+16, ... so the block streams A as one contiguous 8KB/step
// sequence (L2/DRAM friendly). Partials reduced through LDS (fixed order,
// deterministic). 8x resident waves vs wave-per-tile for latency hiding.
// ---------------------------------------------------------------------------
template <int NT>
__global__ void k_matmul(const unsigned short* __restrict__ Asw,
                         const unsigned short* __restrict__ Bsw,
                         float* __restrict__ Y, int M, int nMt, int nKt)
{
    __shared__ float red[8 * NT * 256];
    int mt = blockIdx.x;
    if (mt >= nMt) return;
    int w = threadIdx.x >> 5;
    int lane = threadIdx.x & 31;

    v8f acc[NT];
#pragma unroll
    for (int t = 0; t < NT; ++t)
#pragma unroll
        for (int q = 0; q < 8; ++q) acc[t][q] = 0.0f;

    const unsigned short* Ap = Asw + ((long)mt * nKt + w) * 512 + lane * 16;
    const unsigned short* Bp0 = Bsw + (long)w * NT * 512 + lane * 16;
#pragma unroll 4
    for (int kt = w; kt < nKt; kt += 8) {
        v16bf a = *(const v16bf*)Ap;
        __builtin_prefetch(Ap + 8 * 512, 0, 1);
        const unsigned short* Bp = Bp0;
#pragma unroll
        for (int t = 0; t < NT; ++t) {
            v16bf b = *(const v16bf*)(Bp + t * 512);
            acc[t] = __builtin_amdgcn_wmma_f32_16x16x32_bf16(
                false, a, false, b, (short)0, acc[t], false, false);
        }
        Ap += 8 * 512;
        Bp0 += 8 * NT * 512;
    }

    // deposit partials: element index = row*16+col, row = v + 8*(lane>=16)
    int base = w * (NT * 256);
    int r0 = (lane >= 16) ? 8 : 0;
    int c0 = lane & 15;
#pragma unroll
    for (int t = 0; t < NT; ++t)
#pragma unroll
        for (int v = 0; v < 8; ++v)
            red[base + t * 256 + (r0 + v) * 16 + c0] = acc[t][v];
    __syncthreads();

    // cross-wave reduction (fixed order -> deterministic) + store
    for (int e = threadIdx.x; e < NT * 256; e += 256) {
        float s = 0.0f;
#pragma unroll
        for (int ww = 0; ww < 8; ++ww) s += red[ww * (NT * 256) + e];
        int t = e >> 8, rem = e & 255, r = rem >> 4, c = rem & 15;
        int m = mt * 16 + r;
        if (m < M) Y[(long)m * (NT * 16) + t * 16 + c] = s;
    }
}

// ---------------------------------------------------------------------------
// out[m,o] = act( Y[m,:] . W[o,:] + b[o] + b2[o] ), Cout = 48
// act: 0 none, 1 relu, 2 sigmoid, 3 prelu(alpha[aidx])
// ---------------------------------------------------------------------------
__global__ void k_linear(const float* __restrict__ Y, const float* __restrict__ W,
                         const float* __restrict__ b, const float* __restrict__ b2,
                         float* __restrict__ out, int M, int Cin, int Cout,
                         int act, const float* __restrict__ alpha_p, int aidx)
{
    long t = (long)blockIdx.x * blockDim.x + threadIdx.x;
    if (t >= (long)M * Cout) return;
    int m = (int)(t / Cout), o = (int)(t % Cout);
    float s = b ? b[o] : 0.0f;
    if (b2) s += b2[o];
    const float* yr = Y + (long)m * Cin;
    const float* wr = W + (long)o * Cin;
    for (int c = 0; c < Cin; ++c) s = fmaf(yr[c], wr[c], s);
    if (act == 1) s = s > 0.0f ? s : 0.0f;
    else if (act == 2) s = sigm(s);
    else if (act == 3) { float a = alpha_p[aidx]; s = (s >= 0.0f) ? s : a * s; }
    out[t] = s;
}

// out = sa*a + sb*b + sc*c + sd*d  (null-safe)
__global__ void k_axpy4(float* __restrict__ out,
                        const float* a, float sa, const float* b, float sb,
                        const float* c, float sc, const float* d, float sd, long n)
{
    long i = (long)blockIdx.x * blockDim.x + threadIdx.x;
    if (i >= n) return;
    float s = 0.0f;
    if (a) s += sa * a[i];
    if (b) s += sb * b[i];
    if (c) s += sc * c[i];
    if (d) s += sd * d[i];
    out[i] = s;
}

// out[i,:] = X[g[i],:] * (val ? val[i] : 1)
__global__ void k_gather_scale(float* __restrict__ out, const float* __restrict__ X,
                               const int* __restrict__ g, const float* __restrict__ val,
                               int M, int C)
{
    long t = (long)blockIdx.x * blockDim.x + threadIdx.x;
    if (t >= (long)M * C) return;
    int i = (int)(t / C), c = (int)(t % C);
    float v = X[(long)g[i] * C + c];
    if (val) v *= val[i];
    out[t] = v;
}

__global__ void k_sigmoid(float* __restrict__ out, const float* __restrict__ in, long n)
{
    long i = (long)blockIdx.x * blockDim.x + threadIdx.x;
    if (i < n) out[i] = sigm(in[i]);
}

// sc[m] = sum_c h[row(m),c]*u[m,c] + bp[bidx]; row = gidx ? gidx[m] : m
__global__ void k_rowdot(const float* __restrict__ h, const int* __restrict__ gidx,
                         const float* __restrict__ u, int C,
                         const float* __restrict__ bp, int bidx,
                         float* __restrict__ out, int M)
{
    int m = blockIdx.x * blockDim.x + threadIdx.x;
    if (m >= M) return;
    int r = gidx ? gidx[m] : m;
    const float* hr = h + (long)r * C;
    const float* ur = u + (long)m * C;
    float s = bp[bidx];
    for (int c = 0; c < C; ++c) s = fmaf(hr[c], ur[c], s);
    out[m] = s;
}

__global__ void k_val_sigmoid(float* __restrict__ out, const float* __restrict__ sc,
                              const int* __restrict__ idx, int K)
{
    int i = blockIdx.x * blockDim.x + threadIdx.x;
    if (i < K) out[i] = sigm(sc[idx[i]]);
}

__global__ void k_fill_int(int* __restrict__ p, int v, int n)
{
    int i = blockIdx.x * blockDim.x + threadIdx.x;
    if (i < n) p[i] = v;
}

__global__ void k_scatter_inv(int* __restrict__ inv, const int* __restrict__ idx, int K)
{
    int i = blockIdx.x * blockDim.x + threadIdx.x;
    if (i < K) inv[idx[i]] = i;
}

__global__ void k_compose(int* __restrict__ dst, const int* __restrict__ i1,
                          const int* __restrict__ i2, int n)
{
    int i = blockIdx.x * blockDim.x + threadIdx.x;
    if (i < n) dst[i] = i1[i2[i]];
}

__global__ void k_concat(float* __restrict__ cat, const float* __restrict__ a,
                         const float* __restrict__ b, int M)
{
    long t = (long)blockIdx.x * blockDim.x + threadIdx.x;
    if (t >= (long)M * 96) return;
    int m = (int)(t / 96), c = (int)(t % 96);
    cat[t] = (c < 48) ? a[(long)m * 48 + c] : b[(long)m * 48 + (c - 48)];
}

// ---------------------------------------------------------------------------
// Single-workgroup bitonic sort (descending) of 8192 (key,index) pairs in LDS.
// keys beyond Mreal are -inf so they land at the end.
// ---------------------------------------------------------------------------
#define SORT_N 8192
__global__ void k_sort_desc(const float* __restrict__ keys_in, int Mreal,
                            int* __restrict__ idx_out)
{
    __shared__ float key[SORT_N];
    __shared__ int   val[SORT_N];
    int tid = threadIdx.x;
    for (int i = tid; i < SORT_N; i += blockDim.x) {
        key[i] = (i < Mreal) ? keys_in[i] : -3.0e38f;
        val[i] = i;
    }
    __syncthreads();
    for (int k = 2; k <= SORT_N; k <<= 1) {
        for (int j = k >> 1; j > 0; j >>= 1) {
            for (int i = tid; i < SORT_N; i += blockDim.x) {
                int ixj = i ^ j;
                if (ixj > i) {
                    bool ddir = ((i & k) == 0);     // descending segment
                    float ki = key[i], kj = key[ixj];
                    bool sw = ddir ? (ki < kj) : (ki > kj);
                    if (sw) {
                        key[i] = kj; key[ixj] = ki;
                        int tv = val[i]; val[i] = val[ixj]; val[ixj] = tv;
                    }
                }
            }
            __syncthreads();
        }
    }
    for (int i = tid; i < SORT_N; i += blockDim.x) idx_out[i] = val[i];
}

// ---------------------------------------------------------------------------
// Host orchestration
// ---------------------------------------------------------------------------
extern "C" void kernel_launch(void* const* d_in, const int* in_sizes, int n_in,
                              void* d_out, int out_size, void* d_ws, size_t ws_size,
                              hipStream_t stream)
{
    (void)in_sizes; (void)n_in; (void)out_size; (void)ws_size;

    const int N = 8192, K1 = 7372, K2 = 5160, D = 48;
    const int nMt1 = N / 16, nKt1 = N / 32;                 // 512, 256
    const int nMt2 = (K1 + 15) / 16, nKt2 = (K1 + 31) / 32; // 461, 231
    const int nMt3 = (K2 + 15) / 16, nKt3 = (K2 + 31) / 32; // 323, 162

    const float* A      = (const float*)d_in[0];
    const float* x      = (const float*)d_in[1];
    const int*   perm1  = (const int*)d_in[2];
    const int*   perm2  = (const int*)d_in[3];
    const float* W_in   = (const float*)d_in[4];
    const float* b_in   = (const float*)d_in[5];
    const float* W_dd   = (const float*)d_in[6];
    const float* b_dd   = (const float*)d_in[7];
    const float* W_end  = (const float*)d_in[8];
    const float* b_end  = (const float*)d_in[9];
    const float* mlp_b2 = (const float*)d_in[10];
    const float* preluA = (const float*)d_in[11];
    const float* disc_W = (const float*)d_in[12];
    const float* disc_b = (const float*)d_in[13];

    // ------- output regions -------
    float* outAgg = (float*)d_out;                 // [N,48]
    float* ret1a = outAgg + (long)N * D;           // sc1 scale1 [N]
    float* ret1b = ret1a + N;                      // sc2 scale1 [N]
    float* ret2a = ret1b + N;                      // sc1 scale2 [K1]
    float* ret2b = ret2a + K1;                     // sc2 scale2 [K1]

    // ------- workspace carve -------
    char* wsp = (char*)d_ws;
    size_t off = 0;
    auto carve = [&](size_t bytes) -> void* {
        void* p = wsp + off; off += (bytes + 255) & ~(size_t)255; return p;
    };
    unsigned short* A1sw = (unsigned short*)carve((size_t)nMt1 * nKt1 * 1024);
    unsigned short* A2sw = (unsigned short*)carve((size_t)nMt2 * nKt2 * 1024);
    unsigned short* A3sw = (unsigned short*)carve((size_t)nMt3 * nKt3 * 1024);
    unsigned short* Bsw  = (unsigned short*)carve((size_t)nKt1 * 8 * 1024);
    float* AX = (float*)carve((size_t)N * 128 * 4);
    auto fbuf = [&]() { return (float*)carve((size_t)N * D * 4); };
    float* xs1_0  = fbuf();  // res1 / x_org
    float* h1a    = fbuf();
    float* xdown1 = fbuf();
    float* xcbuf  = fbuf();
    float* ubuf   = fbuf();
    float* xs2_0  = fbuf();  // res2
    float* h1b    = fbuf();
    float* xdown2 = fbuf();
    float* xs3_0  = fbuf();  // res3
    float* t_a = fbuf(); float* t_b = fbuf();
    float* t_c = fbuf(); float* t_d = fbuf();
    float* t_e = fbuf(); float* t_f = fbuf();
    float* x12 = fbuf(); float* x21 = fbuf();
    float* x23 = fbuf(); float* x32 = fbuf();
    float* gtmp = fbuf(); float* sumt = fbuf();
    float* xs2out = fbuf();
    float* cat = (float*)carve((size_t)N * 96 * 4);
    int* idxs1 = (int*)carve(N * 4);
    int* idxs2 = (int*)carve(N * 4);
    int* idx12 = (int*)carve(N * 4);
    int* invm  = (int*)carve(N * 4);
    float* val1 = (float*)carve(N * 4);
    float* val2 = (float*)carve(N * 4);

    auto cdiv = [](long a, long b) { return (int)((a + b - 1) / b); };

    // ------- helpers -------
    auto packA = [&](unsigned short* dst, const int* ridx, const int* cidx,
                     int Mr, int Kc, int nMt, int nKt) {
        long thr = (long)nMt * nKt * 32;
        k_pack_A<<<cdiv(thr, TPB), TPB, 0, stream>>>(A, dst, ridx, cidx, Mr, Kc, nMt, nKt, N);
    };
    auto mm = [&](const unsigned short* Asw, int Mr, int nMt, int nKt,
                  const float* X, int Krows, int C, const int* inv) {
        int nNt = C / 16;
        long thrB = (long)nKt * nNt * 32;
        k_pack_B<<<cdiv(thrB, TPB), TPB, 0, stream>>>(X, Bsw, Krows, C, nKt, nNt, inv);
        if (nNt == 3)      k_matmul<3><<<nMt, TPB, 0, stream>>>(Asw, Bsw, AX, Mr, nMt, nKt);
        else if (nNt == 6) k_matmul<6><<<nMt, TPB, 0, stream>>>(Asw, Bsw, AX, Mr, nMt, nKt);
        else               k_matmul<8><<<nMt, TPB, 0, stream>>>(Asw, Bsw, AX, Mr, nMt, nKt);
    };
    auto lin = [&](const float* Y, const float* W, const float* b, const float* b2,
                   float* out, int M, int Cin, int act, int aidx) {
        long thr = (long)M * D;
        k_linear<<<cdiv(thr, TPB), TPB, 0, stream>>>(Y, W, b, b2, out, M, Cin, D, act, preluA, aidx);
    };
    auto gcn1 = [&](const float* X, const int* inv, int wi, int act, float* out) {
        mm(A1sw, N, nMt1, nKt1, X, N, D, inv);
        lin(AX, W_dd + (long)wi * 2304, b_dd + wi * 48, nullptr, out, N, D, act, 0);
    };
    auto gcn2 = [&](const float* X, const int* inv, int wi, int act, float* out) {
        mm(A2sw, K1, nMt2, nKt2, X, K1, D, inv);
        lin(AX, W_dd + (long)wi * 2304, b_dd + wi * 48, nullptr, out, K1, D, act, 0);
    };
    auto gcn3 = [&](const float* X, int wi, int act, float* out) {
        mm(A3sw, K2, nMt3, nKt3, X, K2, D, nullptr);
        lin(AX, W_dd + (long)wi * 2304, b_dd + wi * 48, nullptr, out, K2, D, act, 0);
    };
    auto axpy = [&](float* out, const float* a, float sa, const float* b, float sb,
                    const float* c, float sc, const float* d, float sd, long n) {
        k_axpy4<<<cdiv(n, TPB), TPB, 0, stream>>>(out, a, sa, b, sb, c, sc, d, sd, n);
    };
    auto make_inv = [&](const int* idx, int K, int Nfull) {
        k_fill_int<<<cdiv(Nfull, TPB), TPB, 0, stream>>>(invm, -1, Nfull);
        k_scatter_inv<<<cdiv(K, TPB), TPB, 0, stream>>>(invm, idx, K);
    };

    // =================== scale construction ===================
    packA(A1sw, nullptr, nullptr, N, N, nMt1, nKt1);

    // x_s1 = gcn(A, x, W_in, b_in)   (C=128)
    mm(A1sw, N, nMt1, nKt1, x, N, 128, nullptr);
    lin(AX, W_in, b_in, nullptr, xs1_0, N, 128, 0, 0);

    // index_select scale 1
    lin(xs1_0, W_dd + 1 * 2304, b_dd + 1 * 48, mlp_b2 + 0, h1a, N, D, 3, 0); // mlp -> h1
    gcn1(h1a, nullptr, 2, 0, xdown1);                                        // hn1 = Xdown_s1
    k_sigmoid<<<cdiv((long)N * D, TPB), TPB, 0, stream>>>(xcbuf, xdown1, (long)N * D);
    lin(xcbuf, disc_W + 0, nullptr, nullptr, ubuf, N, D, 0, 0);              // u = Xc @ Wd^T
    k_rowdot<<<cdiv(N, TPB), TPB, 0, stream>>>(h1a, nullptr, ubuf, D, disc_b, 0, ret1a, N);
    k_rowdot<<<cdiv(N, TPB), TPB, 0, stream>>>(h1a, perm1, ubuf, D, disc_b, 0, ret1b, N);
    k_sort_desc<<<1, 1024, 0, stream>>>(ret1a, N, idxs1);
    k_val_sigmoid<<<cdiv(K1, TPB), TPB, 0, stream>>>(val1, ret1a, idxs1, K1);

    // x_s2 = gcn(A2, x_s1[idx1]*val1, W0)
    k_gather_scale<<<cdiv((long)K1 * D, TPB), TPB, 0, stream>>>(gtmp, xs1_0, idxs1, val1, K1, D);
    packA(A2sw, idxs1, idxs1, K1, K1, nMt2, nKt2);
    gcn2(gtmp, nullptr, 0, 0, xs2_0);

    // index_select scale 2
    lin(xs2_0, W_dd + 3 * 2304, b_dd + 3 * 48, mlp_b2 + 48, h1b, K1, D, 3, 1);
    gcn2(h1b, nullptr, 4, 0, xdown2);                                        // Xdown_s2
    k_sigmoid<<<cdiv((long)K1 * D, TPB), TPB, 0, stream>>>(xcbuf, xdown2, (long)K1 * D);
    lin(xcbuf, disc_W + 2304, nullptr, nullptr, ubuf, K1, D, 0, 0);
    k_rowdot<<<cdiv(K1, TPB), TPB, 0, stream>>>(h1b, nullptr, ubuf, D, disc_b, 1, ret2a, K1);
    k_rowdot<<<cdiv(K1, TPB), TPB, 0, stream>>>(h1b, perm2, ubuf, D, disc_b, 1, ret2b, K1);
    k_sort_desc<<<1, 1024, 0, stream>>>(ret2a, K1, idxs2);
    k_val_sigmoid<<<cdiv(K2, TPB), TPB, 0, stream>>>(val2, ret2a, idxs2, K2);

    k_gather_scale<<<cdiv((long)K2 * D, TPB), TPB, 0, stream>>>(xs3_0, xs2_0, idxs2, val2, K2, D);
    k_compose<<<cdiv(K2, TPB), TPB, 0, stream>>>(idx12, idxs1, idxs2, K2);
    packA(A3sw, idx12, idx12, K2, K2, nMt3, nKt3);

    // =================== layer 1 ===================
    gcn1(xs1_0, nullptr, 5, 1, t_a);
    gcn2(xs2_0, nullptr, 8, 1, t_c);
    gcn3(xs3_0, 11, 1, t_e);
    // cross 1
    gcn1(t_a, nullptr, 14, 0, gtmp);
    k_gather_scale<<<cdiv((long)K1 * D, TPB), TPB, 0, stream>>>(x12, gtmp, idxs1, val1, K1, D);
    make_inv(idxs1, K1, N);
    gcn1(t_c, invm, 15, 0, x21);
    gcn2(t_c, nullptr, 16, 0, gtmp);
    k_gather_scale<<<cdiv((long)K2 * D, TPB), TPB, 0, stream>>>(x23, gtmp, idxs2, val2, K2, D);
    make_inv(idxs2, K2, K1);
    gcn2(t_e, invm, 17, 0, x32);
    axpy(t_b, t_a, 1.f, x21, 1.f, xs1_0, 1.f, nullptr, 0.f, (long)N * D);
    axpy(t_d, t_c, 1.f, x12, 0.5f, x32, 0.5f, xs2_0, 1.f, (long)K1 * D);
    axpy(t_f, t_e, 1.f, x23, 1.f, xs3_0, 1.f, nullptr, 0.f, (long)K2 * D);

    // =================== layer 2 ===================
    gcn1(t_b, nullptr, 6, 1, t_a);
    gcn2(t_d, nullptr, 9, 1, t_c);
    gcn3(t_f, 12, 1, t_e);
    // cross 2 (scaled 0.05)
    gcn1(t_a, nullptr, 18, 0, gtmp);
    k_gather_scale<<<cdiv((long)K1 * D, TPB), TPB, 0, stream>>>(x12, gtmp, idxs1, val1, K1, D);
    make_inv(idxs1, K1, N);
    gcn1(t_c, invm, 19, 0, x21);
    gcn2(t_c, nullptr, 20, 0, gtmp);
    k_gather_scale<<<cdiv((long)K2 * D, TPB), TPB, 0, stream>>>(x23, gtmp, idxs2, val2, K2, D);
    make_inv(idxs2, K2, K1);
    gcn2(t_e, invm, 21, 0, x32);
    axpy(t_b, t_a, 1.f, x21, 0.05f, nullptr, 0.f, nullptr, 0.f, (long)N * D);
    axpy(t_d, t_c, 1.f, x12, 0.025f, x32, 0.025f, nullptr, 0.f, (long)K1 * D);
    axpy(t_f, t_e, 1.f, x23, 0.05f, nullptr, 0.f, nullptr, 0.f, (long)K2 * D);

    // =================== layer 3 ===================
    gcn1(t_b, nullptr, 7, 1, t_a);
    gcn2(t_d, nullptr, 10, 1, t_c);
    gcn3(t_f, 13, 1, t_e);

    // =================== fuse back ===================
    // x_s3_out = unpool(A2, x_s3, idx2, 23) + Xdown_s2
    make_inv(idxs2, K2, K1);
    gcn2(t_e, invm, 23, 0, gtmp);
    axpy(x32, gtmp, 1.f, xdown2, 1.f, nullptr, 0.f, nullptr, 0.f, (long)K1 * D); // x_s3_out
    // x_s2_out = unpool(A, x_s2 + x_s3_out, idx1, 22)
    axpy(sumt, t_c, 1.f, x32, 1.f, nullptr, 0.f, nullptr, 0.f, (long)K1 * D);
    make_inv(idxs1, K1, N);
    gcn1(sumt, invm, 22, 0, xs2out);
    // x_agg = x_s1 + x_s2_out + Xdown_s1
    axpy(gtmp, t_a, 1.f, xs2out, 1.f, xdown1, 1.f, nullptr, 0.f, (long)N * D);
    // concat + final gcn with W_end
    k_concat<<<cdiv((long)N * 96, TPB), TPB, 0, stream>>>(cat, gtmp, xs1_0, N);
    mm(A1sw, N, nMt1, nKt1, cat, N, 96, nullptr);
    lin(AX, W_end, b_end, nullptr, outAgg, N, 96, 0, 0);
}
